// SelfAttention_62500364091558
// MI455X (gfx1250) — matrix operand order repaired and checked by reading, not compiled, not run on hardware
//
#include <hip/hip_runtime.h>

// ---------------------------------------------------------------------------
// Problem constants (from the reference): B=4, T=2048, D=1024, H=16, S=64
// ---------------------------------------------------------------------------
constexpr int cB = 4;
constexpr int cT = 2048;
constexpr int cD = 1024;
constexpr int cH = 16;
constexpr int cS = 64;          // head dim
constexpr int cM = cB * cT;     // 8192 rows for the projections

typedef __bf16 bhalf;
typedef __attribute__((ext_vector_type(16))) __bf16 v16bf;
typedef __attribute__((ext_vector_type(8)))  __bf16 v8bf;
typedef __attribute__((ext_vector_type(4)))  __bf16 v4bf;
typedef __attribute__((ext_vector_type(8)))  float  v8f;

union bfrag { v16bf v; v8bf h[2]; };

static __device__ __forceinline__ v8f wmma_bf16(const bfrag& a, const bfrag& b, v8f c) {
    return __builtin_amdgcn_wmma_f32_16x16x32_bf16(
        /*neg_a=*/false, a.v, /*neg_b=*/false, b.v,
        /*c_mod=*/(short)0, c, /*reuse_a=*/false, /*reuse_b=*/false);
}

// Scheduling fence: keep the fragment-load block from being sunk into the
// WMMA block (forces all ds_loads in flight at once -> staggered dscnt waits
// instead of s_wait_dscnt 0x0 before every v_wmma).
static __device__ __forceinline__ void sched_fence() {
#if __has_builtin(__builtin_amdgcn_sched_barrier)
    __builtin_amdgcn_sched_barrier(0);
#endif
}

// ---------------------------------------------------------------------------
// CDNA5 async global->LDS copy (ASYNCcnt-tracked, no VGPR round trip).
// dst operand = LDS byte offset (low 32 bits of the generic pointer).
// ---------------------------------------------------------------------------
static __device__ __forceinline__ void async_copy_b128(void* lds, const void* gptr) {
    unsigned ldsOff = (unsigned)(size_t)lds;
    asm volatile("global_load_async_to_lds_b128 %0, %1, off"
                 :: "v"(ldsOff), "v"(gptr) : "memory");
}
static __device__ __forceinline__ void wait_async0() {
#if __has_builtin(__builtin_amdgcn_s_wait_asynccnt)
    __builtin_amdgcn_s_wait_asynccnt(0);
#else
    asm volatile("s_wait_asynccnt 0x0" ::: "memory");
#endif
}

// ---------------------------------------------------------------------------
// 16-lane butterfly reductions via DPP16 (VALU, no LDS, no dscnt stalls).
// ---------------------------------------------------------------------------
template <int CTRL>
static __device__ __forceinline__ float dpp_movf(float x) {
    const int xi = __builtin_bit_cast(int, x);
    return __builtin_bit_cast(float,
        __builtin_amdgcn_update_dpp(xi, xi, CTRL, 0xF, 0xF, true));
}
static __device__ __forceinline__ float red16_max(float x) {
    x = fmaxf(x, dpp_movf<0xB1>(x));    // quad_perm(1,0,3,2)
    x = fmaxf(x, dpp_movf<0x4E>(x));    // quad_perm(2,3,0,1)
    x = fmaxf(x, dpp_movf<0x141>(x));   // row_half_mirror
    x = fmaxf(x, dpp_movf<0x140>(x));   // row_mirror
    return x;
}
static __device__ __forceinline__ float red16_sum(float x) {
    x += dpp_movf<0xB1>(x);
    x += dpp_movf<0x4E>(x);
    x += dpp_movf<0x141>(x);
    x += dpp_movf<0x140>(x);
    return x;
}

static __device__ __forceinline__ v4bf pack4(float4 v) {
    v4bf p;
    p[0] = (bhalf)v.x; p[1] = (bhalf)v.y; p[2] = (bhalf)v.z; p[3] = (bhalf)v.w;
    return p;
}

// ---------------------------------------------------------------------------
// Tile staging for the GEMM (one 128x32 A tile + one 128x32 W tile).
// f32 A: convert + packed ds_store_b64.  bf16 A: async global->LDS b128.
// ---------------------------------------------------------------------------
template <typename AT>
static __device__ __forceinline__ void stage_tiles(
    bhalf (* __restrict__ As)[40], bhalf (* __restrict__ Bs)[40],
    const AT* __restrict__ aRow, const float* __restrict__ wRow,
    int k0, int tid)
{
    if constexpr (sizeof(AT) == 4) {
        const float4* src = (const float4*)(aRow + k0);
        #pragma unroll
        for (int i = 0; i < 8; ++i)
            *(v4bf*)&As[tid][4*i] = pack4(src[i]);       // ds_store_b64
    } else {
        #pragma unroll
        for (int i = 0; i < 4; ++i)
            async_copy_b128(&As[tid][8*i], aRow + k0 + 8*i);
    }
    const float4* src = (const float4*)(wRow + k0);
    #pragma unroll
    for (int i = 0; i < 8; ++i)
        *(v4bf*)&Bs[tid][4*i] = pack4(src[i]);
}

// ---------------------------------------------------------------------------
// GEMM: C[M,N] = A[M,K] @ W[N,K]^T (+ bias).  128x128 block tile, 64x64 wave
// tile, K-step 32, ping-pong LDS (one barrier per K-step).
// ---------------------------------------------------------------------------
template <typename AT, typename OT>
__global__ __launch_bounds__(128) void gemm_tn_bf16(
    const AT* __restrict__ A, const float* __restrict__ W,
    const float* __restrict__ bias, OT* __restrict__ C,
    int M, int N, int K)
{
    __shared__ bhalf As[2][128][40];   // 80B row stride (16B aligned)
    __shared__ bhalf Bs[2][128][40];

    const int tid  = threadIdx.x;
    const int wave = tid >> 5;
    const int lane = tid & 31;
    const int half = lane >> 4;
    const int l16  = lane & 15;
    const int wm   = (wave >> 1) * 64;
    const int wn   = (wave & 1) * 64;
    const int m0   = blockIdx.x * 128;
    const int n0   = blockIdx.y * 128;

    const AT*    aRow = A + (size_t)(m0 + tid) * K;
    const float* wRow = W + (size_t)(n0 + tid) * K;

    v8f acc[4][4] = {};

    stage_tiles<AT>(As[0], Bs[0], aRow, wRow, 0, tid);

    int buf = 0;
    for (int k0 = 0; k0 < K; k0 += 32, buf ^= 1) {
        if constexpr (sizeof(AT) == 2) wait_async0();    // finish copies into `buf`
        __syncthreads();                                 // publish `buf`

        if (k0 + 32 < K) {
            __builtin_prefetch(aRow + k0 + 64, 0, 1);
            __builtin_prefetch(wRow + k0 + 64, 0, 1);
            stage_tiles<AT>(As[buf ^ 1], Bs[buf ^ 1], aRow, wRow, k0 + 32, tid);
        }

        bfrag af[4], bf_[4];
        #pragma unroll
        for (int i = 0; i < 4; ++i) {
            const bhalf* p = &As[buf][wm + 16*i + l16][0];
            af[i].h[0] = *(const v8bf*)(p + 8*half);
            af[i].h[1] = *(const v8bf*)(p + 16 + 8*half);
        }
        #pragma unroll
        for (int j = 0; j < 4; ++j) {
            const bhalf* p = &Bs[buf][wn + 16*j + l16][0];
            bf_[j].h[0] = *(const v8bf*)(p + 8*half);
            bf_[j].h[1] = *(const v8bf*)(p + 16 + 8*half);
        }
        sched_fence();   // all 16 ds_loads issued before any wmma
        #pragma unroll
        for (int i = 0; i < 4; ++i)
            #pragma unroll
            for (int j = 0; j < 4; ++j)
                acc[i][j] = wmma_bf16(af[i], bf_[j], acc[i][j]);
    }

    // ---- epilogue: C layout lane L -> col l16, rows v + 8*half ----
    #pragma unroll
    for (int j = 0; j < 4; ++j) {
        const int col = n0 + wn + 16*j + l16;
        float bv = 0.f;
        if (bias) bv = bias[col];
        #pragma unroll
        for (int i = 0; i < 4; ++i) {
            #pragma unroll
            for (int v = 0; v < 8; ++v) {
                const int row = m0 + wm + 16*i + v + 8*half;
                if constexpr (sizeof(OT) == 4)
                    C[(size_t)row * N + col] = acc[i][j][v] + bv;
                else
                    C[(size_t)row * N + col] = (OT)acc[i][j][v];
            }
        }
    }
}

// ---------------------------------------------------------------------------
// Flash attention over Q (=K=V per the reference bug), bf16 in / bf16 out.
// Grid: (T/64, H, B).  4 waves; each wave owns 16 query rows.
// ---------------------------------------------------------------------------
__global__ __launch_bounds__(128) void attn_flash_bf16(
    const bhalf* __restrict__ Q, bhalf* __restrict__ O)
{
    __shared__ bhalf Kv[64][72];     // [kv][s]  144B row stride (16B aligned)
    __shared__ bhalf KvT[64][72];    // [s][kv]  transposed copy for P@V B-frags
    __shared__ bhalf Pl[4][16][72];  // per-wave P relayout C->A form

    const int tid  = threadIdx.x;
    const int wave = tid >> 5;
    const int lane = tid & 31;
    const int half = lane >> 4;
    const int l16  = lane & 15;
    const int b    = blockIdx.z;
    const int h    = blockIdx.y;
    const int q0   = blockIdx.x * 64 + wave * 16;

    const bhalf* Qbh = Q + (size_t)b * cT * cD + (size_t)h * cS;

    // Q fragments (A-matrix form), kept in registers across the KV loop.
    bfrag qf[2];
    {
        const bhalf* qrow = Qbh + (size_t)(q0 + l16) * cD;
        #pragma unroll
        for (int ks = 0; ks < 2; ++ks) {
            qf[ks].h[0] = *(const v8bf*)(qrow + ks*32 + 8*half);
            qf[ks].h[1] = *(const v8bf*)(qrow + ks*32 + 16 + 8*half);
        }
    }

    v8f   Oacc[4] = {};
    float rowM[8], rowL[8];
    #pragma unroll
    for (int v = 0; v < 8; ++v) { rowM[v] = -3.0e38f; rowL[v] = 0.f; }

    const int rStage = tid >> 1;
    const int cStage = (tid & 1) * 32;

    for (int kv0 = 0; kv0 < cT; kv0 += 64) {
        // ---- stage 64x64 KV tile (and its transpose) into LDS ----
        {
            const bhalf* src = Qbh + (size_t)(kv0 + rStage) * cD + cStage;
            if (kv0 + 64 < cT)
                __builtin_prefetch(src + (size_t)64 * cD, 0, 1);
            v8bf t0 = ((const v8bf*)src)[0];
            v8bf t1 = ((const v8bf*)src)[1];
            v8bf t2 = ((const v8bf*)src)[2];
            v8bf t3 = ((const v8bf*)src)[3];
            *(v8bf*)&Kv[rStage][cStage + 0]  = t0;
            *(v8bf*)&Kv[rStage][cStage + 8]  = t1;
            *(v8bf*)&Kv[rStage][cStage + 16] = t2;
            *(v8bf*)&Kv[rStage][cStage + 24] = t3;
            #pragma unroll
            for (int i = 0; i < 8; ++i) {
                KvT[cStage + 0  + i][rStage] = t0[i];
                KvT[cStage + 8  + i][rStage] = t1[i];
                KvT[cStage + 16 + i][rStage] = t2[i];
                KvT[cStage + 24 + i][rStage] = t3[i];
            }
        }
        __syncthreads();

        // ---- S = Q K^T  (16 x 64 per wave), batched frag loads ----
        v8f sc[4] = {};
        #pragma unroll
        for (int ks = 0; ks < 2; ++ks) {
            bfrag kb[4];
            #pragma unroll
            for (int j = 0; j < 4; ++j) {
                const bhalf* p = &Kv[j*16 + l16][ks*32 + 8*half];
                kb[j].h[0] = *(const v8bf*)p;
                kb[j].h[1] = *(const v8bf*)(p + 16);
            }
            sched_fence();   // keep 8 ds_loads in flight ahead of the wmmas
            #pragma unroll
            for (int j = 0; j < 4; ++j)
                sc[j] = wmma_bf16(qf[ks], kb[j], sc[j]);
        }

        // ---- online softmax (scale = 1/sqrt(64) = 0.125), DPP reductions ----
        float corr[8];
        #pragma unroll
        for (int v = 0; v < 8; ++v) {
            float m = fmaxf(fmaxf(sc[0][v], sc[1][v]), fmaxf(sc[2][v], sc[3][v]));
            m = red16_max(m) * 0.125f;
            const float mn = fmaxf(rowM[v], m);
            const float cr = __expf(rowM[v] - mn);
            rowM[v] = mn;
            float ls = 0.f;
            #pragma unroll
            for (int j = 0; j < 4; ++j) {
                float p = __expf(sc[j][v] * 0.125f - mn);
                sc[j][v] = p;
                ls += p;
            }
            ls = red16_sum(ls);
            rowL[v] = rowL[v] * cr + ls;
            corr[v] = cr;
        }
        #pragma unroll
        for (int j = 0; j < 4; ++j)
            #pragma unroll
            for (int v = 0; v < 8; ++v)
                Oacc[j][v] *= corr[v];

        // ---- relayout P (C-form) -> LDS -> A-form bf16 frags ----
        #pragma unroll
        for (int j = 0; j < 4; ++j)
            #pragma unroll
            for (int v = 0; v < 8; ++v)
                Pl[wave][v + 8*half][j*16 + l16] = (bhalf)sc[j][v];
        // per-wave LDS region: hardware DS waits order the reads below

        // ---- O += P @ V, batched frag loads ----
        #pragma unroll
        for (int ks = 0; ks < 2; ++ks) {
            bfrag pa, vb[4];
            const bhalf* pp = &Pl[wave][l16][0];
            pa.h[0] = *(const v8bf*)(pp + ks*32 + 8*half);
            pa.h[1] = *(const v8bf*)(pp + ks*32 + 16 + 8*half);
            #pragma unroll
            for (int j = 0; j < 4; ++j) {
                const bhalf* p = &KvT[j*16 + l16][ks*32 + 8*half];
                vb[j].h[0] = *(const v8bf*)p;
                vb[j].h[1] = *(const v8bf*)(p + 16);
            }
            sched_fence();   // keep 10 ds_loads in flight ahead of the wmmas
            #pragma unroll
            for (int j = 0; j < 4; ++j)
                Oacc[j] = wmma_bf16(pa, vb[j], Oacc[j]);
        }
        __syncthreads();
    }

    // ---- normalize and store (merge heads layout: (B,T,D), d = h*64 + s) ----
    bhalf* Obh = O + (size_t)b * cT * cD + (size_t)h * cS;
    float inv[8];
    #pragma unroll
    for (int v = 0; v < 8; ++v) inv[v] = 1.0f / rowL[v];
    #pragma unroll
    for (int j = 0; j < 4; ++j) {
        #pragma unroll
        for (int v = 0; v < 8; ++v) {
            const int t = q0 + v + 8*half;
            Obh[(size_t)t * cD + j*16 + l16] = (bhalf)(Oacc[j][v] * inv[v]);
        }
    }
}

// ---------------------------------------------------------------------------
// Launch: Q = x Wq^T (bf16 -> ws) ; flash-attn (bf16 ws -> bf16 ws) ;
//         out = attn Wu^T + bu (f32 -> d_out)
// ---------------------------------------------------------------------------
extern "C" void kernel_launch(void* const* d_in, const int* in_sizes, int n_in,
                              void* d_out, int out_size, void* d_ws, size_t ws_size,
                              hipStream_t stream) {
    (void)in_sizes; (void)n_in; (void)out_size; (void)ws_size;
    const float* x  = (const float*)d_in[0];
    const float* Wq = (const float*)d_in[1];
    const float* Wu = (const float*)d_in[2];
    const float* bu = (const float*)d_in[3];
    float* out = (float*)d_out;

    bhalf* Qb = (bhalf*)d_ws;                        // 8192*1024 bf16 = 16 MB
    bhalf* Ab = Qb + (size_t)cM * cD;                // 16 MB more

    dim3 gGemm(cM / 128, cD / 128);
    gemm_tn_bf16<float, bhalf><<<gGemm, 128, 0, stream>>>(x, Wq, nullptr, Qb, cM, cD, cD);

    dim3 gAttn(cT / 64, cH, cB);
    attn_flash_bf16<<<gAttn, 128, 0, stream>>>(Qb, Ab);

    gemm_tn_bf16<bhalf, float><<<gGemm, 128, 0, stream>>>(Ab, Wu, bu, out, cM, cD, cD);
}
